// CellContentlAttention_10892037063175
// MI455X (gfx1250) — matrix operand order repaired and checked by reading, not compiled, not run on hardware
//
#include <hip/hip_runtime.h>
#include <hip/hip_bf16.h>

#define ENC    512
#define STR_H  256
#define CELL_H 512
#define ATTN   512
#define BB     64
#define LL     1024

typedef __attribute__((ext_vector_type(16))) __bf16 v16bf;
typedef __attribute__((ext_vector_type(8)))  float  v8f;

// ---------------------------------------------------------------------------
// Convert + transpose W_enc[e][a] (f32) -> Wt[a][e] (bf16) so B-fragments are
// contiguous along K.
// ---------------------------------------------------------------------------
__global__ void __launch_bounds__(256) wconv_kernel(const float* __restrict__ W,
                                                    __bf16* __restrict__ Wt) {
    int idx = blockIdx.x * 256 + threadIdx.x;   // over 512*512
    int a = idx >> 9;
    int e = idx & 511;
    Wt[(size_t)a * ENC + e] = (__bf16)W[(size_t)e * ATTN + a];
}

// ---------------------------------------------------------------------------
// bias_total[b][a] = str[b]@W_str + b_str + cell[b]@W_cell + b_cell + b_enc
// ---------------------------------------------------------------------------
__global__ void __launch_bounds__(256) bias_kernel(
    const float* __restrict__ str, const float* __restrict__ cell,
    const float* __restrict__ Wstr, const float* __restrict__ bstr,
    const float* __restrict__ Wcell, const float* __restrict__ bcell,
    const float* __restrict__ benc, float* __restrict__ bias_total) {
    int b = blockIdx.x;
    int t = threadIdx.x;
    __shared__ float s[STR_H + CELL_H];
    for (int i = t; i < STR_H; i += 256)  s[i] = str[b * STR_H + i];
    for (int i = t; i < CELL_H; i += 256) s[STR_H + i] = cell[b * CELL_H + i];
    __syncthreads();
    for (int a = t; a < ATTN; a += 256) {
        float acc = bstr[a] + bcell[a] + benc[a];
        for (int k = 0; k < STR_H; ++k)  acc += s[k] * Wstr[k * ATTN + a];
        for (int k = 0; k < CELL_H; ++k) acc += s[STR_H + k] * Wcell[k * ATTN + a];
        bias_total[b * ATTN + a] = acc;
    }
}

// ---------------------------------------------------------------------------
// Main WMMA kernel: one workgroup per (b, 64-row L tile).  Four 16-row M
// sub-tiles share every B fragment -> 4x less W_enc (L2) traffic.
// Unrolling is explicitly bounded to keep VGPR pressure < 256 (no spills).
// scores[b][l] = relu(X[b,l,:]@W_enc + bias_total[b,:]) . W_comb
// ---------------------------------------------------------------------------
#define XS_STRIDE 520   // 512 + 8 bf16 pad -> conflict-free b128 LDS reads
#define MROWS     64    // rows per workgroup
#define MSUB      4     // 16-row sub-tiles

__global__ void __launch_bounds__(256) score_kernel(
    const float* __restrict__ X, const __bf16* __restrict__ Wt,
    const float* __restrict__ bias_total, const float* __restrict__ Wcomb,
    const float* __restrict__ bcomb, float* __restrict__ scores) {
    int wg   = blockIdx.x;
    int b    = wg >> 4;                 // L/64 = 16 tiles per batch
    int m0   = (wg & 15) << 6;
    int t    = threadIdx.x;
    int wave = t >> 5;
    int lane = t & 31;

    __shared__ __bf16 Xs[MROWS * XS_STRIDE];   // ~65 KB
    __shared__ float  sc[MROWS];

    // Stage 64x512 fp32 tile -> bf16 LDS (float4 loads, packed 8B LDS stores)
    const float* Xb = X + ((size_t)b * LL + m0) * ENC;
    #pragma unroll 4
    for (int j = 0; j < 32; ++j) {
        int v   = j * 256 + t;              // float4 index 0..8191
        int row = v >> 7;                   // 128 float4 per row
        int e   = (v & 127) << 2;
        float4 x4 = *(const float4*)(Xb + (size_t)row * ENC + e);
        union { __bf16 h[4]; uint2 u; } p;
        p.h[0] = (__bf16)x4.x; p.h[1] = (__bf16)x4.y;
        p.h[2] = (__bf16)x4.z; p.h[3] = (__bf16)x4.w;
        *(uint2*)&Xs[row * XS_STRIDE + e] = p.u;
    }
    for (int i = t; i < MROWS; i += 256) sc[i] = 0.0f;
    __syncthreads();

    // ISA 16-bit A layout: lanes 0-15 M=lane, K runs {0-7,16-23}; lanes 16-31
    // same M, K runs {8-15,24-31}.  B layout: same K runs as A, N = lane&15.
    int row = lane & 15;
    int ka  = (lane >> 4) << 3;   // A-fragment K sub-offset (0 or 8)
    int kb  = (lane >> 4) << 3;   // B-fragment K sub-offset (0 or 8)
    float partial[MSUB][8];
    #pragma unroll
    for (int s = 0; s < MSUB; ++s)
        #pragma unroll
        for (int r = 0; r < 8; ++r) partial[s][r] = 0.0f;

    #pragma unroll 1
    for (int tile = 0; tile < 4; ++tile) {
        int n = ((wave + tile * 8) << 4) + (lane & 15);
        const __bf16* wrow = Wt + (size_t)n * ENC;
        v8f acc[MSUB];
        #pragma unroll
        for (int s = 0; s < MSUB; ++s) acc[s] = (v8f){};

        #pragma unroll 2
        for (int k0 = 0; k0 < ENC; k0 += 32) {
            union { v16bf v; uint4 q[2]; } Bf;
            Bf.q[0] = *(const uint4*)&wrow[k0 + kb];
            Bf.q[1] = *(const uint4*)&wrow[k0 + kb + 16];
            #pragma unroll
            for (int s = 0; s < MSUB; ++s) {
                union { v16bf v; uint4 q[2]; } A;
                int arow = (s << 4) + row;
                A.q[0] = *(const uint4*)&Xs[arow * XS_STRIDE + k0 + ka];
                A.q[1] = *(const uint4*)&Xs[arow * XS_STRIDE + k0 + ka + 16];
                acc[s] = __builtin_amdgcn_wmma_f32_16x16x32_bf16(
                    false, A.v, false, Bf.v, (short)0, acc[s], false, false);
            }
        }
        // Fused epilogue: + bias, ReLU, dot with W_comb column
        float bias = bias_total[b * ATTN + n];
        float wc   = Wcomb[n];
        #pragma unroll
        for (int s = 0; s < MSUB; ++s)
            #pragma unroll
            for (int r = 0; r < 8; ++r) {
                float v = acc[s][r] + bias;
                v = v > 0.0f ? v : 0.0f;
                partial[s][r] += v * wc;
            }
    }
    // C/D layout: lanes 0-15 -> M=r, lanes 16-31 -> M=r+8. Reduce over N lanes.
    int mrow = (lane >> 4) << 3;
    #pragma unroll
    for (int s = 0; s < MSUB; ++s)
        #pragma unroll
        for (int r = 0; r < 8; ++r)
            atomicAdd(&sc[(s << 4) + mrow + r], partial[s][r]);
    __syncthreads();
    if (t < MROWS) scores[b * LL + m0 + t] = sc[t] + bcomb[0];
}

// ---------------------------------------------------------------------------
// Softmax over L per batch -> normalized weights
// ---------------------------------------------------------------------------
__global__ void __launch_bounds__(256) softmax_kernel(
    const float* __restrict__ scores, float* __restrict__ wts) {
    int b = blockIdx.x;
    int t = threadIdx.x;
    __shared__ float w[LL];
    __shared__ float red[256];

    float m = -3.4e38f;
    #pragma unroll
    for (int j = 0; j < 4; ++j) {
        float s = scores[b * LL + j * 256 + t];
        w[j * 256 + t] = s;
        m = fmaxf(m, s);
    }
    red[t] = m;
    __syncthreads();
    for (int s = 128; s > 0; s >>= 1) {
        if (t < s) red[t] = fmaxf(red[t], red[t + s]);
        __syncthreads();
    }
    float mx = red[0];
    __syncthreads();

    float sum = 0.0f;
    #pragma unroll
    for (int j = 0; j < 4; ++j) {
        float e = __expf(w[j * 256 + t] - mx);
        w[j * 256 + t] = e;
        sum += e;
    }
    red[t] = sum;
    __syncthreads();
    for (int s = 128; s > 0; s >>= 1) {
        if (t < s) red[t] += red[t + s];
        __syncthreads();
    }
    float inv = 1.0f / red[0];
    #pragma unroll
    for (int j = 0; j < 4; ++j) wts[b * LL + j * 256 + t] = w[j * 256 + t] * inv;
}

__global__ void __launch_bounds__(256) zero_kernel(float* __restrict__ out, int n) {
    int i = blockIdx.x * 256 + threadIdx.x;
    if (i < n) out[i] = 0.0f;
}

// ---------------------------------------------------------------------------
// context[b][e] = sum_l wts[b][l] * X[b][l][e]; L split 8-way per batch,
// partials combined with global_atomic_add_f32.
// ---------------------------------------------------------------------------
__global__ void __launch_bounds__(256) context_kernel(
    const float* __restrict__ X, const float* __restrict__ wts,
    float* __restrict__ out) {
    int blk = blockIdx.x;
    int b   = blk >> 3;
    int l0  = (blk & 7) * 128;
    int t   = threadIdx.x;
    int e0  = t * 2;

    __shared__ float wl[128];
    if (t < 128) wl[t] = wts[b * LL + l0 + t];
    __syncthreads();

    const float* Xb = X + ((size_t)b * LL + l0) * ENC;
    float2 acc = {0.0f, 0.0f};
    for (int l = 0; l < 128; ++l) {
        float2 x = *(const float2*)(Xb + (size_t)l * ENC + e0);
        float w  = wl[l];
        acc.x += w * x.x;
        acc.y += w * x.y;
    }
    atomicAdd(&out[b * ENC + e0], acc.x);
    atomicAdd(&out[b * ENC + e0 + 1], acc.y);
}

// ---------------------------------------------------------------------------
extern "C" void kernel_launch(void* const* d_in, const int* in_sizes, int n_in,
                              void* d_out, int out_size, void* d_ws, size_t ws_size,
                              hipStream_t stream) {
    const float* X      = (const float*)d_in[0];   // [B, L, ENC]
    const float* str    = (const float*)d_in[1];   // [1, B, STR_H]
    const float* cell   = (const float*)d_in[2];   // [1, B, CELL_H]
    const float* W_enc  = (const float*)d_in[3];   // [ENC, ATTN]
    const float* b_enc  = (const float*)d_in[4];
    const float* W_str  = (const float*)d_in[5];
    const float* b_str  = (const float*)d_in[6];
    const float* W_cell = (const float*)d_in[7];
    const float* b_cell = (const float*)d_in[8];
    const float* W_comb = (const float*)d_in[9];   // [ATTN, 1]
    const float* b_comb = (const float*)d_in[10];
    float* out = (float*)d_out;                    // [B, ENC]

    char* ws = (char*)d_ws;
    __bf16* Wt        = (__bf16*)(ws);                       // 512KB
    float*  bias_tot  = (float*)(ws + 524288);               // 128KB
    float*  scores    = (float*)(ws + 524288 + 131072);      // 256KB
    float*  wts       = (float*)(ws + 524288 + 131072 + 262144); // 256KB

    wconv_kernel<<<(ENC * ATTN) / 256, 256, 0, stream>>>(W_enc, Wt);
    bias_kernel<<<BB, 256, 0, stream>>>(str, cell, W_str, b_str, W_cell, b_cell,
                                        b_enc, bias_tot);
    score_kernel<<<BB * (LL / MROWS), 256, 0, stream>>>(X, Wt, bias_tot, W_comb,
                                                        b_comb, scores);
    softmax_kernel<<<BB, 256, 0, stream>>>(scores, wts);
    zero_kernel<<<(BB * ENC + 255) / 256, 256, 0, stream>>>(out, BB * ENC);
    context_kernel<<<BB * 8, 256, 0, stream>>>(X, wts, out);
}
